// MultiDilatelocalAttention_63883343561395
// MI455X (gfx1250) — matrix operand order repaired and checked by reading, not compiled, not run on hardware
//
#include <hip/hip_runtime.h>
#include <hip/hip_bf16.h>

// ---------------------------------------------------------------------------
// Multi-dilated local attention for MI455X (gfx1250, wave32, WMMA).
//   x:(4,4096,768)  w_qkv:(2304,768)  w_proj:(768,768)  b_proj:(768,)
// Outputs (concat): proj_out (B,N,C) ; v,k,q each (3,B,256,N)
// GEMMs: v_wmma_f32_16x16x32_f16, 128x64 block tile, 2x2 wave tile.
// ---------------------------------------------------------------------------

typedef _Float16 f16;
typedef __attribute__((ext_vector_type(16))) _Float16 v16h;
typedef __attribute__((ext_vector_type(8)))  _Float16 v8h;
typedef __attribute__((ext_vector_type(8)))  float    v8f;

#define DIM      768
#define NSEQ     4096
#define BATCH    4
#define QKV_COLS 2304
#define SEG      (BATCH * NSEQ * DIM)   /* 12,582,912 floats per output segment */

#define BK   32      // K-step per WMMA
#define LDT  40      // padded LDS row stride in halves (80 B, keeps 16B align)

#define WMMA(A, B, C) \
  __builtin_amdgcn_wmma_f32_16x16x32_f16(false, (A), false, (B), (short)0, (C), false, false)

// ---- WMMA fragment gathers (per ISA 7.12.2 wave32 layouts) -----------------
// A 16x32 f16: lanes 0-15 -> M=lane, K=0..7 (elems 0-7) and K=16..23 (8-15);
//              lanes 16-31 -> same M, K=8..15 and K=24..31.
__device__ __forceinline__ v16h frag_a(const f16* smem, int rowBase, int lane) {
  int m = lane & 15, kh = lane >> 4;
  const f16* p = smem + (rowBase + m) * LDT + kh * 8;
  v8h lo = *(const v8h*)(p);        // K = kh*8 + 0..7
  v8h hi = *(const v8h*)(p + 16);   // K = 16 + kh*8 + 0..7
  v16h a;
  #pragma unroll
  for (int e = 0; e < 8; ++e) { a[e] = lo[e]; a[8 + e] = hi[e]; }
  return a;
}
// B 32x16 f16 (stored transposed in LDS as [n][k]):
//   lanes 0-15 -> N=lane, K=0..7,16..23 ; lanes 16-31 -> N=lane-16, K=8..15,24..31.
__device__ __forceinline__ v16h frag_b(const f16* smem, int colBase, int lane) {
  int n = lane & 15, kh = lane >> 4;
  const f16* p = smem + (colBase + n) * LDT + kh * 8;
  v8h lo = *(const v8h*)(p);
  v8h hi = *(const v8h*)(p + 16);
  v16h b;
  #pragma unroll
  for (int e = 0; e < 8; ++e) { b[e] = lo[e]; b[8 + e] = hi[e]; }
  return b;
}

// ---------------------------------------------------------------------------
// Kernel 1: qkv = x @ w_qkv.T, scattered into (nd,B,256,N) q/k/v segments.
// Block = 128(M) x 64(N); 8 waves as 4(mi) x 2(nj); wave tile = 32x32 (2x2).
// ---------------------------------------------------------------------------
__global__ __launch_bounds__(256)
void qkv_gemm(const float* __restrict__ x, const float* __restrict__ wqkv,
              float* __restrict__ out) {
  __shared__ f16 lA[128 * LDT];  // x tile [m][k]
  __shared__ f16 lB[64 * LDT];   // w tile [n][k]  (B transposed)
  const int tid  = threadIdx.x;
  const int lane = tid & 31, wave = tid >> 5;
  const int mi = wave & 3;            // 4 m-slabs of 32
  const int nj = wave >> 2;           // 2 n-slabs of 32
  const int row0 = blockIdx.y * 128;
  const int col0 = blockIdx.x * 64;

  const int arow = tid >> 1;          // 0..127 : A tile row
  const int acol = (tid & 1) * 16;    // 0/16   : A k offset (16 floats each)
  const int brow = tid >> 2;          // 0..63  : B tile row
  const int bcol = (tid & 3) * 8;     // 0,8,16,24

  v8f c00 = {}, c01 = {}, c10 = {}, c11 = {};

  for (int k0 = 0; k0 < DIM; k0 += BK) {
    __syncthreads();
    {
      const float* srcA = x + (size_t)(row0 + arow) * DIM + k0 + acol;
      f16* dstA = lA + arow * LDT + acol;
      #pragma unroll
      for (int j = 0; j < 16; ++j) dstA[j] = (f16)srcA[j];
      const float* srcB = wqkv + (size_t)(col0 + brow) * DIM + k0 + bcol;
      f16* dstB = lB + brow * LDT + bcol;
      #pragma unroll
      for (int j = 0; j < 8; ++j) dstB[j] = (f16)srcB[j];
    }
    __syncthreads();
    v16h a0 = frag_a(lA, mi * 32,      lane);
    v16h a1 = frag_a(lA, mi * 32 + 16, lane);
    v16h b0 = frag_b(lB, nj * 32,      lane);
    v16h b1 = frag_b(lB, nj * 32 + 16, lane);
    c00 = WMMA(a0, b0, c00);
    c01 = WMMA(a0, b1, c01);
    c10 = WMMA(a1, b0, c10);
    c11 = WMMA(a1, b1, c11);
  }

  // C layout: lane<16 -> (M=r, N=lane) ; lane>=16 -> (M=8+r, N=lane-16)
  const int nlane = lane & 15;
  const int mbase = (lane >> 4) * 8;
  #pragma unroll
  for (int tm = 0; tm < 2; ++tm) {
    #pragma unroll
    for (int tn = 0; tn < 2; ++tn) {
      v8f acc = tm ? (tn ? c11 : c10) : (tn ? c01 : c00);
      const int gcol = col0 + nj * 32 + tn * 16 + nlane;     // 0..2303
      const int s   = gcol / DIM;                            // 0=q 1=k 2=v
      const int rem = gcol - s * DIM;
      const int ig  = rem >> 8;                              // dilation group
      const int cch = rem & 255;                             // channel in group
      float* seg = out + (size_t)(3 - s) * SEG;              // ret order: v,k,q
      #pragma unroll
      for (int r = 0; r < 8; ++r) {
        const int row = row0 + mi * 32 + tm * 16 + mbase + r;
        const int b  = row >> 12;
        const int nn = row & (NSEQ - 1);
        seg[(((size_t)((ig * BATCH + b) * 256 + cch)) << 12) + nn] = acc[r];
      }
    }
  }
}

// ---------------------------------------------------------------------------
// Kernel 2: 3-tap dilated local attention with the reference's 9-slot softmax
// (6 implicit zero logits -> denominator + 6; zero-padded edges give score 0).
// Reads q/k/v from the (nd,B,256,N) segments (n-contiguous, coalesced).
// Writes attention output in (B, C, N) scratch layout for the proj GEMM.
// ---------------------------------------------------------------------------
__global__ __launch_bounds__(256)
void dilated_attn(const float* __restrict__ qkv, float* __restrict__ attn_ws) {
  const int tid = threadIdx.x;
  const int n   = blockIdx.x * 256 + tid;
  const int y   = blockIdx.y;                 // 48 = 3 groups * 4 batch * 4 heads
  const int ig  = y >> 4;
  const int rem = y & 15;
  const int b   = rem >> 2;
  const int h   = rem & 3;
  const int dil = ig + 1;

  const float* q = qkv + (size_t)3 * SEG;
  const float* k = qkv + (size_t)2 * SEG;
  const float* v = qkv + (size_t)1 * SEG;
  const size_t cb = ((size_t)((ig * BATCH + b) * 256 + h * 64)) << 12;

  const int nm = n - dil, np = n + dil;
  const bool vm = (nm >= 0), vp = (np < NSEQ);

  float s0 = 0.f, s1 = 0.f, s2 = 0.f;
  #pragma unroll 4
  for (int d = 0; d < 64; ++d) {
    const size_t off = cb + ((size_t)d << 12);
    const float qd = q[off + n];
    s1 += qd * k[off + n];
    if (vm) s0 += qd * k[off + nm];
    if (vp) s2 += qd * k[off + np];
  }
  const float scale = 0.125f;                 // 64^-0.5
  const float e0 = __expf(s0 * scale);
  const float e1 = __expf(s1 * scale);
  const float e2 = __expf(s2 * scale);
  const float inv = 1.0f / (e0 + e1 + e2 + 6.0f);   // 6 zero logits
  const float p0 = vm ? e0 * inv : 0.f;
  const float p1 = e1 * inv;
  const float p2 = vp ? e2 * inv : 0.f;

  float* dst = attn_ws + ((size_t)(b * DIM + ig * 256 + h * 64) << 12) + n;
  #pragma unroll 4
  for (int d = 0; d < 64; ++d) {
    const size_t off = cb + ((size_t)d << 12);
    float o = p1 * v[off + n];
    if (vm) o += p0 * v[off + nm];
    if (vp) o += p2 * v[off + np];
    dst[(size_t)d << 12] = o;
  }
}

// ---------------------------------------------------------------------------
// Kernel 3: out0 = attn @ w_proj.T + b_proj.
// A comes from (B,C,N) scratch -> transposed into LDS [m][k] (coalesced reads).
// ---------------------------------------------------------------------------
__global__ __launch_bounds__(256)
void proj_gemm(const float* __restrict__ attn_ws, const float* __restrict__ wproj,
               const float* __restrict__ bias, float* __restrict__ out0) {
  __shared__ f16 lA[128 * LDT];
  __shared__ f16 lB[64 * LDT];
  const int tid  = threadIdx.x;
  const int lane = tid & 31, wave = tid >> 5;
  const int mi = wave & 3;
  const int nj = wave >> 2;
  const int row0 = blockIdx.y * 128;
  const int col0 = blockIdx.x * 64;
  const int bb = row0 >> 12;
  const int n0 = row0 & (NSEQ - 1);           // 128-row tile stays in one batch

  const int ac  = tid >> 3;                   // 0..31 : k (channel) for A
  const int an  = (tid & 7) * 16;             // 0..112 : n offset for A
  const int brow = tid >> 2;                  // B tile row
  const int bcol = (tid & 3) * 8;

  v8f c00 = {}, c01 = {}, c10 = {}, c11 = {};

  for (int k0 = 0; k0 < DIM; k0 += BK) {
    __syncthreads();
    {
      const float* srcA = attn_ws + ((size_t)(bb * DIM + k0 + ac) << 12) + n0 + an;
      #pragma unroll
      for (int j = 0; j < 16; ++j) lA[(an + j) * LDT + ac] = (f16)srcA[j];
      const float* srcB = wproj + (size_t)(col0 + brow) * DIM + k0 + bcol;
      f16* dstB = lB + brow * LDT + bcol;
      #pragma unroll
      for (int j = 0; j < 8; ++j) dstB[j] = (f16)srcB[j];
    }
    __syncthreads();
    v16h a0 = frag_a(lA, mi * 32,      lane);
    v16h a1 = frag_a(lA, mi * 32 + 16, lane);
    v16h b0 = frag_b(lB, nj * 32,      lane);
    v16h b1 = frag_b(lB, nj * 32 + 16, lane);
    c00 = WMMA(a0, b0, c00);
    c01 = WMMA(a0, b1, c01);
    c10 = WMMA(a1, b0, c10);
    c11 = WMMA(a1, b1, c11);
  }

  const int nlane = lane & 15;
  const int mbase = (lane >> 4) * 8;
  #pragma unroll
  for (int tm = 0; tm < 2; ++tm) {
    #pragma unroll
    for (int tn = 0; tn < 2; ++tn) {
      v8f acc = tm ? (tn ? c11 : c10) : (tn ? c01 : c00);
      const int gcol = col0 + nj * 32 + tn * 16 + nlane;
      const float bv = bias[gcol];
      #pragma unroll
      for (int r = 0; r < 8; ++r) {
        const int row = row0 + mi * 32 + tm * 16 + mbase + r;
        out0[(size_t)row * DIM + gcol] = acc[r] + bv;
      }
    }
  }
}

// ---------------------------------------------------------------------------
extern "C" void kernel_launch(void* const* d_in, const int* in_sizes, int n_in,
                              void* d_out, int out_size, void* d_ws, size_t ws_size,
                              hipStream_t stream) {
  const float* x     = (const float*)d_in[0];
  const float* wqkv  = (const float*)d_in[1];
  const float* wproj = (const float*)d_in[2];
  const float* bproj = (const float*)d_in[3];
  float* out = (float*)d_out;
  float* ws  = (float*)d_ws;    // needs 4*768*4096*4 = 50,331,648 bytes

  dim3 blk(256);
  qkv_gemm   <<<dim3(QKV_COLS / 64, (BATCH * NSEQ) / 128), blk, 0, stream>>>(x, wqkv, out);
  dilated_attn<<<dim3(NSEQ / 256, 48),                     blk, 0, stream>>>(out, ws);
  proj_gemm  <<<dim3(DIM / 64, (BATCH * NSEQ) / 128),      blk, 0, stream>>>(ws, wproj, bproj, out);
}
